// LinearAverage_24352464569137
// MI455X (gfx1250) — matrix operand order repaired and checked by reading, not compiled, not run on hardware
//
#include <hip/hip_runtime.h>

typedef __bf16  v16bf __attribute__((ext_vector_type(16)));
typedef __bf16  v8bf  __attribute__((ext_vector_type(8)));
typedef __bf16  v4bf  __attribute__((ext_vector_type(4)));
typedef float   v8f   __attribute__((ext_vector_type(8)));
typedef float   v4f   __attribute__((ext_vector_type(4)));

#define DDIM    128     // embedding dim (K)
#define LPAD    8       // LDS row pad: stride 136 bf16 = 272B = 68 banks -> conflict-free
#define LSTRIDE (DDIM + LPAD)
#define NT      32      // memory rows (output cols) per block
#define THREADS 512     // 16 waves; wave w -> x rows [16w, 16w+16)

// ---------------------------------------------------------------------------
// Prologue: split x (fp32) into bf16 hi/lo pair once (L2-resident afterwards).
// hi = bf16(f); lo = bf16(f - f32(hi))  ->  f ~= hi + lo to ~2^-16 relative.
// ---------------------------------------------------------------------------
__global__ void split_x_kernel(const float* __restrict__ x,
                               __bf16* __restrict__ xh,
                               __bf16* __restrict__ xl, int n) {
  int i = blockIdx.x * blockDim.x + threadIdx.x;
  if (i < n) {
    float f = x[i];
    __bf16 h = (__bf16)f;
    xh[i] = h;
    xl[i] = (__bf16)(f - (float)h);
  }
}

// ---------------------------------------------------------------------------
// Main GEMM: out[m, n] = dot(x[m,:], memory[n,:]) * (1/T)
// 3-pass bf16 emulation of fp32: acc += Ah*Bh + Ah*Bl + Al*Bh
// ---------------------------------------------------------------------------
template <bool PRESPLIT>
__global__ __launch_bounds__(THREADS, 1)
void simgemm_kernel(const float* __restrict__ x,
                    const __bf16* __restrict__ xh,
                    const __bf16* __restrict__ xl,
                    const float* __restrict__ memory,
                    const float* __restrict__ params,
                    float* __restrict__ out, int N) {
  __shared__ __align__(32) __bf16 sBh[NT][LSTRIDE];
  __shared__ __align__(32) __bf16 sBl[NT][LSTRIDE];

  const int tid  = threadIdx.x;
  const int lane = tid & 31;
  const int wave = tid >> 5;
  const int hseg = (lane >> 4) & 1;   // 0: lanes 0-15, 1: lanes 16-31
  const int l15  = lane & 15;

  const long  nBase = (long)blockIdx.x * NT;
  const float invT  = 1.0f / params[0];

  // ---- cooperative load + hi/lo split of B tile (memory rows nBase..nBase+31)
  {
#pragma unroll
    for (int j = 0; j < (NT * DDIM) / (THREADS * 4); ++j) {
      int e   = (tid + j * THREADS) * 4;
      int row = e >> 7;          // / DDIM
      int col = e & (DDIM - 1);
      v4f v = *(const v4f*)(memory + (nBase + row) * DDIM + col);
      v4bf h, l;
#pragma unroll
      for (int q = 0; q < 4; ++q) {
        __bf16 hq = (__bf16)v[q];
        h[q] = hq;
        l[q] = (__bf16)(v[q] - (float)hq);
      }
      *(v4bf*)&sBh[row][col] = h;
      *(v4bf*)&sBl[row][col] = l;
    }
  }

  // ---- A fragments for full K into registers (ISA 16-bit A 16x32 layout):
  // lane<16: K = k+0..7 & k+16..23 ; lane>=16: K = k+8..15 & k+24..31
  const int mRow = wave * 16 + l15;
  v16bf Ah[4], Al[4];
#pragma unroll
  for (int k = 0; k < 4; ++k) {
    const int c0 = k * 32 + hseg * 8;
    if (PRESPLIT) {
      v8bf h0 = *(const v8bf*)(xh + mRow * DDIM + c0);
      v8bf h1 = *(const v8bf*)(xh + mRow * DDIM + c0 + 16);
      v8bf l0 = *(const v8bf*)(xl + mRow * DDIM + c0);
      v8bf l1 = *(const v8bf*)(xl + mRow * DDIM + c0 + 16);
      Ah[k] = __builtin_shufflevector(h0, h1, 0,1,2,3,4,5,6,7,8,9,10,11,12,13,14,15);
      Al[k] = __builtin_shufflevector(l0, l1, 0,1,2,3,4,5,6,7,8,9,10,11,12,13,14,15);
    } else {
      v4f f0 = *(const v4f*)(x + mRow * DDIM + c0);
      v4f f1 = *(const v4f*)(x + mRow * DDIM + c0 + 4);
      v4f f2 = *(const v4f*)(x + mRow * DDIM + c0 + 16);
      v4f f3 = *(const v4f*)(x + mRow * DDIM + c0 + 20);
      float f[16] = {f0[0],f0[1],f0[2],f0[3], f1[0],f1[1],f1[2],f1[3],
                     f2[0],f2[1],f2[2],f2[3], f3[0],f3[1],f3[2],f3[3]};
#pragma unroll
      for (int i = 0; i < 16; ++i) {
        __bf16 h = (__bf16)f[i];
        Ah[k][i] = h;
        Al[k][i] = (__bf16)(f[i] - (float)h);
      }
    }
  }

  __syncthreads();

  // ---- WMMA main loop: 2 column tiles x 4 K-steps x 3 passes = 24 WMMA/wave
  v8f acc[NT / 16] = {};
#pragma unroll
  for (int t = 0; t < NT / 16; ++t) {
    const int bRow = t * 16 + l15;   // B column n = lane&15 (+16t)
#pragma unroll
    for (int k = 0; k < 4; ++k) {
      // ISA 16-bit B 32x16 layout: lane<16: K=k+0..7 & k+16..23 ; lane>=16: K=k+8..15 & k+24..31
      const int cb = k * 32 + hseg * 8;
      const __bf16* hp = &sBh[bRow][cb];
      const __bf16* lp = &sBl[bRow][cb];
      v8bf bh0 = *(const v8bf*)hp;
      v8bf bh1 = *(const v8bf*)(hp + 16);
      v8bf bl0 = *(const v8bf*)lp;
      v8bf bl1 = *(const v8bf*)(lp + 16);
      v16bf bh = __builtin_shufflevector(bh0, bh1, 0,1,2,3,4,5,6,7,8,9,10,11,12,13,14,15);
      v16bf bl = __builtin_shufflevector(bl0, bl1, 0,1,2,3,4,5,6,7,8,9,10,11,12,13,14,15);
      acc[t] = __builtin_amdgcn_wmma_f32_16x16x32_bf16(false, Ah[k], false, bh,
                                                       (short)0, acc[t], false, false);
      acc[t] = __builtin_amdgcn_wmma_f32_16x16x32_bf16(false, Ah[k], false, bl,
                                                       (short)0, acc[t], false, false);
      acc[t] = __builtin_amdgcn_wmma_f32_16x16x32_bf16(false, Al[k], false, bh,
                                                       (short)0, acc[t], false, false);
    }
  }

  // ---- scale by 1/T and store (C layout: lane=N, VGPR r = M row, +8 for hseg)
#pragma unroll
  for (int t = 0; t < NT / 16; ++t) {
    const long col = nBase + t * 16 + l15;
#pragma unroll
    for (int r = 0; r < 8; ++r) {
      const int m = wave * 16 + r + hseg * 8;
      out[(long)m * N + col] = acc[t][r] * invT;
    }
  }
}

// ---------------------------------------------------------------------------
extern "C" void kernel_launch(void* const* d_in, const int* in_sizes, int n_in,
                              void* d_out, int out_size, void* d_ws, size_t ws_size,
                              hipStream_t stream) {
  const float* x      = (const float*)d_in[0];
  // d_in[1] = y (indices): unused by the forward similarity output
  const float* memory = (const float*)d_in[2];
  const float* params = (const float*)d_in[3];
  float* out = (float*)d_out;

  const int xElems = in_sizes[0];        // 256*128
  const int N      = in_sizes[2] / DDIM; // 500000

  const size_t splitBytes = (size_t)xElems * sizeof(__bf16) * 2;
  const bool presplit = (d_ws != nullptr) && (ws_size >= splitBytes);

  dim3 grid(N / NT);
  dim3 block(THREADS);

  if (presplit) {
    __bf16* xh = (__bf16*)d_ws;
    __bf16* xl = xh + xElems;
    split_x_kernel<<<(xElems + 255) / 256, 256, 0, stream>>>(x, xh, xl, xElems);
    simgemm_kernel<true><<<grid, block, 0, stream>>>(x, xh, xl, memory, params, out, N);
  } else {
    simgemm_kernel<false><<<grid, block, 0, stream>>>(x, nullptr, nullptr, memory, params, out, N);
  }
}
